// LASErMPNN_Decoder_67877663146008
// MI455X (gfx1250) — matrix operand order repaired and checked by reading, not compiled, not run on hardware
//
#include <hip/hip_runtime.h>

typedef __attribute__((ext_vector_type(16))) __bf16 bhalf16;
typedef __attribute__((ext_vector_type(8)))  float  float8;

#define N_PROT  8192
#define N_LIG   2048
#define E_PP_N  163840
#define E_LP_N  81920
#define HD      128
#define NHEAD   4
#define D_FF    512

union AF { bhalf16 v; __bf16 h[16]; };
union BW { uint4 u[2]; bhalf16 v; };
union CF { float8 v; float f[8]; };

__device__ __forceinline__ float geluf(float x) {
  float t = tanhf(0.7978845608028654f * (x + 0.044715f * x * x * x));
  return 0.5f * x * (1.0f + t);
}
__device__ __forceinline__ float lreluf(float x) { return x > 0.f ? x : 0.2f * x; }
__device__ __forceinline__ unsigned f2ord(float f) {
  unsigned u = __float_as_uint(f);
  return (u & 0x80000000u) ? ~u : (u | 0x80000000u);
}
__device__ __forceinline__ float ord2f(unsigned u) {
  unsigned b = (u & 0x80000000u) ? (u & 0x7FFFFFFFu) : ~u;
  return __uint_as_float(b);
}

// ---------------------------------------------------------------- prep kernels

__global__ void wprep_kernel(const float* __restrict__ src, __bf16* __restrict__ dst,
                             int K, int N) {
  int i = blockIdx.x * blockDim.x + threadIdx.x;
  if (i >= K * N) return;
  int k = i / N, n = i - k * N;
  dst[(size_t)n * K + k] = (__bf16)src[i];
}

__global__ void init_ws_kernel(float* __restrict__ attnb, float* __restrict__ sumb,
                               unsigned* __restrict__ maxb) {
  int i = blockIdx.x * blockDim.x + threadIdx.x;
  if (i < N_PROT * HD) attnb[i] = 0.f;
  if (i < N_PROT * NHEAD) { sumb[i] = 0.f; maxb[i] = f2ord(-1e30f); }
}

// ------------------------------------------------------------- edge GAT kernel
// One wave handles a 16-edge tile: D = z(16xK) @ W(Kx128) for both att and val.
// z gathered on the fly; mode 0: [src_edge_feats(448) || prot_sink(128)],
// mode 1: [lig_src(128) || prot_sink(128) || eattr(128)].

__device__ __forceinline__ const float* z_base(int mode, int k, int e, int src, int snk,
                                               const float* featA, const float* prot,
                                               const float* eattr) {
  if (mode == 0) {
    return (k < 448) ? (featA + (size_t)e * 448 + k)
                     : (prot + (size_t)snk * HD + (k - 448));
  }
  if (k < 128) return featA + (size_t)src * HD + k;
  if (k < 256) return prot + (size_t)snk * HD + (k - 128);
  return eattr + (size_t)e * HD + (k - 256);
}

__global__ __launch_bounds__(128) void edge_gat_kernel(
    const float* __restrict__ featA, const float* __restrict__ prot,
    const float* __restrict__ eattr, const int* __restrict__ eidx,
    const __bf16* __restrict__ WattT, const float* __restrict__ batt,
    const float* __restrict__ avec,
    const __bf16* __restrict__ WvalT, const float* __restrict__ bval,
    __bf16* __restrict__ valout, float* __restrict__ logout,
    unsigned* __restrict__ maxbuf, int E, int K, int mode) {
  int tile = blockIdx.x * 4 + (threadIdx.x >> 5);
  int e0 = tile * 16;
  if (e0 >= E) return;
  int lane  = threadIdx.x & 31;
  int lrow  = lane & 15;
  int ho    = (lane >> 4) * 8;    // unused; A and B both use khi
  int khi   = (lane >> 4) * 16;   // B-matrix K half-offset
  int mbase = (lane >> 4) * 8;    // C rows handled by this half-wave

  int myE  = e0 + lrow;
  int msrc = eidx[myE];
  int msnk = eidx[E + myE];

  float8 za[8], zv[8];
#pragma unroll
  for (int t = 0; t < 8; ++t) {
    za[t] = (float8){0, 0, 0, 0, 0, 0, 0, 0};
    zv[t] = (float8){0, 0, 0, 0, 0, 0, 0, 0};
  }

  for (int k0 = 0; k0 < K; k0 += 32) {
    AF A;
    const float* p1 = z_base(mode, k0 + khi,     myE, msrc, msnk, featA, prot, eattr);
    const float* p2 = z_base(mode, k0 + 8 + khi, myE, msrc, msnk, featA, prot, eattr);
#pragma unroll
    for (int j = 0; j < 8; ++j) A.h[j]     = (__bf16)p1[j];
#pragma unroll
    for (int j = 0; j < 8; ++j) A.h[8 + j] = (__bf16)p2[j];
    int kst = k0 + khi;
#pragma unroll
    for (int t = 0; t < 8; ++t) {
      int n = t * 16 + lrow;
      BW Ba;
      const uint4* qa = (const uint4*)(WattT + (size_t)n * K + kst);
      Ba.u[0] = qa[0]; Ba.u[1] = qa[1];
      za[t] = __builtin_amdgcn_wmma_f32_16x16x32_bf16(false, A.v, false, Ba.v,
                                                      (short)0, za[t], false, false);
      BW Bv;
      const uint4* qv = (const uint4*)(WvalT + (size_t)n * K + kst);
      Bv.u[0] = qv[0]; Bv.u[1] = qv[1];
      zv[t] = __builtin_amdgcn_wmma_f32_16x16x32_bf16(false, A.v, false, Bv.v,
                                                      (short)0, zv[t], false, false);
    }
  }

  // Bias, emit bf16 values, keep biased att for logits.
  float attv[8][8];
#pragma unroll
  for (int t = 0; t < 8; ++t) {
    int n = t * 16 + lrow;
    float ba = batt[n], bv = bval[n];
    CF ca, cv;
    ca.v = za[t]; cv.v = zv[t];
#pragma unroll
    for (int r = 0; r < 8; ++r) {
      attv[t][r] = ca.f[r] + ba;
      int e = e0 + mbase + r;
      valout[(size_t)e * HD + n] = (__bf16)(cv.f[r] + bv);
    }
  }

  // GATv2 logits: leaky_relu(att) . a per head; butterfly-reduce the 16 lanes
  // of each half-wave (xor masks 1/2/4/8 never cross the 16-lane boundary).
#pragma unroll
  for (int h = 0; h < NHEAD; ++h) {
    float aw1 = avec[h * 32 + lrow];
    float aw2 = avec[h * 32 + 16 + lrow];
    float p[8];
#pragma unroll
    for (int r = 0; r < 8; ++r)
      p[r] = lreluf(attv[2 * h][r]) * aw1 + lreluf(attv[2 * h + 1][r]) * aw2;
#pragma unroll
    for (int off = 1; off < 16; off <<= 1) {
#pragma unroll
      for (int r = 0; r < 8; ++r) p[r] += __shfl_xor(p[r], off, 32);
    }
    if (lrow == 0) {
#pragma unroll
      for (int r = 0; r < 8; ++r) {
        int e = e0 + mbase + r;
        int snk = eidx[E + e];
        logout[(size_t)e * NHEAD + h] = p[r];
        atomicMax(&maxbuf[snk * NHEAD + h], f2ord(p[r]));
      }
    }
  }
}

// ----------------------------------------------------- segment softmax kernels

__global__ void softmax_pass_kernel(const int* __restrict__ eidx_pp,
                                    const int* __restrict__ eidx_lp,
                                    float* __restrict__ logitb,
                                    const unsigned* __restrict__ maxb,
                                    float* __restrict__ sumb) {
  int gid = blockIdx.x * blockDim.x + threadIdx.x;
  const int tot = (E_PP_N + E_LP_N) * NHEAD;
  if (gid >= tot) return;
  int e = gid >> 2, h = gid & 3;
  int snk = (e < E_PP_N) ? eidx_pp[E_PP_N + e] : eidx_lp[E_LP_N + (e - E_PP_N)];
  float m  = ord2f(maxb[snk * NHEAD + h]);
  float ev = __expf(logitb[gid] - m);
  logitb[gid] = ev;
  atomicAdd(&sumb[snk * NHEAD + h], ev);
}

__global__ __launch_bounds__(128) void aggregate_kernel(
    const int* __restrict__ eidx_pp, const int* __restrict__ eidx_lp,
    const float* __restrict__ ebuf, const float* __restrict__ sumb,
    const __bf16* __restrict__ valb, float* __restrict__ attnb) {
  int e = blockIdx.x;
  int n = threadIdx.x;
  int snk = (e < E_PP_N) ? eidx_pp[E_PP_N + e] : eidx_lp[E_LP_N + (e - E_PP_N)];
  int h = n >> 5;
  float alpha = ebuf[(size_t)e * NHEAD + h] / (sumb[snk * NHEAD + h] + 1e-9f);
  float v = (float)valb[(size_t)e * HD + n];
  atomicAdd(&attnb[(size_t)snk * HD + n], v * alpha);
}

// -------------------------------------------------------- node update kernel
// One wave per 16-node tile: attn -> head MLP -> LN -> FF MLP -> LN. LDS used
// to re-layout C tiles into A-matrix operand layout between GEMMs.

__global__ __launch_bounds__(32) void node_update_kernel(
    const float* __restrict__ attnb, const float* __restrict__ prot,
    const __bf16* __restrict__ Wh1T, const float* __restrict__ bh1,
    const __bf16* __restrict__ Wh2T, const float* __restrict__ bh2,
    const float* __restrict__ g1, const float* __restrict__ b1,
    const __bf16* __restrict__ Wm1T, const float* __restrict__ bm1,
    const __bf16* __restrict__ Wm2T, const float* __restrict__ bm2,
    const float* __restrict__ g2, const float* __restrict__ b2,
    float* __restrict__ out) {
  __shared__ float xbuf[16 * HD];
  __shared__ float ffbuf[16 * D_FF];
  int lane  = threadIdx.x & 31;
  int lrow  = lane & 15;
  int ho    = (lane >> 4) * 8;
  int khi   = (lane >> 4) * 16;
  int mbase = (lane >> 4) * 8;
  int n0    = blockIdx.x * 16;

  float8 acc[8];

  // ---- Stage 1: t1 = gelu(attn @ Wh1 + bh1) -> xbuf ----
#pragma unroll
  for (int t = 0; t < 8; ++t) acc[t] = (float8){0, 0, 0, 0, 0, 0, 0, 0};
  for (int k0 = 0; k0 < HD; k0 += 32) {
    AF A;
    const float* p1 = attnb + (size_t)(n0 + lrow) * HD + k0 + khi;
#pragma unroll
    for (int j = 0; j < 8; ++j) A.h[j]     = (__bf16)p1[j];
#pragma unroll
    for (int j = 0; j < 8; ++j) A.h[8 + j] = (__bf16)p1[8 + j];
    int kst = k0 + khi;
#pragma unroll
    for (int t = 0; t < 8; ++t) {
      int n = t * 16 + lrow;
      BW B;
      const uint4* q = (const uint4*)(Wh1T + (size_t)n * HD + kst);
      B.u[0] = q[0]; B.u[1] = q[1];
      acc[t] = __builtin_amdgcn_wmma_f32_16x16x32_bf16(false, A.v, false, B.v,
                                                       (short)0, acc[t], false, false);
    }
  }
#pragma unroll
  for (int t = 0; t < 8; ++t) {
    int n = t * 16 + lrow;
    CF c; c.v = acc[t];
    float bb = bh1[n];
#pragma unroll
    for (int r = 0; r < 8; ++r) xbuf[(mbase + r) * HD + n] = geluf(c.f[r] + bb);
  }
  __syncthreads();

  // ---- Stage 2: x = LN(prot + t1 @ Wh2 + bh2) ----
#pragma unroll
  for (int t = 0; t < 8; ++t) acc[t] = (float8){0, 0, 0, 0, 0, 0, 0, 0};
  for (int k0 = 0; k0 < HD; k0 += 32) {
    AF A;
    const float* p1 = &xbuf[lrow * HD + k0 + khi];
#pragma unroll
    for (int j = 0; j < 8; ++j) A.h[j]     = (__bf16)p1[j];
#pragma unroll
    for (int j = 0; j < 8; ++j) A.h[8 + j] = (__bf16)p1[8 + j];
    int kst = k0 + khi;
#pragma unroll
    for (int t = 0; t < 8; ++t) {
      int n = t * 16 + lrow;
      BW B;
      const uint4* q = (const uint4*)(Wh2T + (size_t)n * HD + kst);
      B.u[0] = q[0]; B.u[1] = q[1];
      acc[t] = __builtin_amdgcn_wmma_f32_16x16x32_bf16(false, A.v, false, B.v,
                                                       (short)0, acc[t], false, false);
    }
  }
  float xv[8][8];
#pragma unroll
  for (int t = 0; t < 8; ++t) {
    int n = t * 16 + lrow;
    CF c; c.v = acc[t];
    float bb = bh2[n];
#pragma unroll
    for (int r = 0; r < 8; ++r)
      xv[t][r] = c.f[r] + bb + prot[(size_t)(n0 + mbase + r) * HD + n];
  }
  {
    float s[8], ss[8];
#pragma unroll
    for (int r = 0; r < 8; ++r) {
      s[r] = 0.f; ss[r] = 0.f;
#pragma unroll
      for (int t = 0; t < 8; ++t) { s[r] += xv[t][r]; ss[r] += xv[t][r] * xv[t][r]; }
    }
#pragma unroll
    for (int off = 1; off < 16; off <<= 1) {
#pragma unroll
      for (int r = 0; r < 8; ++r) {
        s[r]  += __shfl_xor(s[r], off, 32);
        ss[r] += __shfl_xor(ss[r], off, 32);
      }
    }
    float mu[8], rs[8];
#pragma unroll
    for (int r = 0; r < 8; ++r) {
      mu[r] = s[r] * (1.f / HD);
      float var = ss[r] * (1.f / HD) - mu[r] * mu[r];
      rs[r] = rsqrtf(var + 1e-5f);
    }
#pragma unroll
    for (int t = 0; t < 8; ++t) {
      int n = t * 16 + lrow;
      float gg = g1[n], bb = b1[n];
#pragma unroll
      for (int r = 0; r < 8; ++r) xv[t][r] = (xv[t][r] - mu[r]) * rs[r] * gg + bb;
    }
  }
  __syncthreads();
#pragma unroll
  for (int t = 0; t < 8; ++t) {
    int n = t * 16 + lrow;
#pragma unroll
    for (int r = 0; r < 8; ++r) xbuf[(mbase + r) * HD + n] = xv[t][r];
  }
  __syncthreads();

  // ---- Stage 3: ff = gelu(x @ Wm1 + bm1) -> ffbuf (4 column chunks) ----
  for (int cc = 0; cc < 4; ++cc) {
#pragma unroll
    for (int t = 0; t < 8; ++t) acc[t] = (float8){0, 0, 0, 0, 0, 0, 0, 0};
    for (int k0 = 0; k0 < HD; k0 += 32) {
      AF A;
      const float* p1 = &xbuf[lrow * HD + k0 + khi];
#pragma unroll
      for (int j = 0; j < 8; ++j) A.h[j]     = (__bf16)p1[j];
#pragma unroll
      for (int j = 0; j < 8; ++j) A.h[8 + j] = (__bf16)p1[8 + j];
      int kst = k0 + khi;
#pragma unroll
      for (int t = 0; t < 8; ++t) {
        int n = cc * 128 + t * 16 + lrow;
        BW B;
        const uint4* q = (const uint4*)(Wm1T + (size_t)n * HD + kst);
        B.u[0] = q[0]; B.u[1] = q[1];
        acc[t] = __builtin_amdgcn_wmma_f32_16x16x32_bf16(false, A.v, false, B.v,
                                                         (short)0, acc[t], false, false);
      }
    }
#pragma unroll
    for (int t = 0; t < 8; ++t) {
      int n = cc * 128 + t * 16 + lrow;
      CF c; c.v = acc[t];
      float bb = bm1[n];
#pragma unroll
      for (int r = 0; r < 8; ++r) ffbuf[(mbase + r) * D_FF + n] = geluf(c.f[r] + bb);
    }
  }
  __syncthreads();

  // ---- Stage 4: out = LN(x + ff @ Wm2 + bm2) ----
#pragma unroll
  for (int t = 0; t < 8; ++t) acc[t] = (float8){0, 0, 0, 0, 0, 0, 0, 0};
  for (int k0 = 0; k0 < D_FF; k0 += 32) {
    AF A;
    const float* p1 = &ffbuf[lrow * D_FF + k0 + khi];
#pragma unroll
    for (int j = 0; j < 8; ++j) A.h[j]     = (__bf16)p1[j];
#pragma unroll
    for (int j = 0; j < 8; ++j) A.h[8 + j] = (__bf16)p1[8 + j];
    int kst = k0 + khi;
#pragma unroll
    for (int t = 0; t < 8; ++t) {
      int n = t * 16 + lrow;
      BW B;
      const uint4* q = (const uint4*)(Wm2T + (size_t)n * D_FF + kst);
      B.u[0] = q[0]; B.u[1] = q[1];
      acc[t] = __builtin_amdgcn_wmma_f32_16x16x32_bf16(false, A.v, false, B.v,
                                                       (short)0, acc[t], false, false);
    }
  }
  float yv[8][8];
#pragma unroll
  for (int t = 0; t < 8; ++t) {
    int n = t * 16 + lrow;
    CF c; c.v = acc[t];
    float bb = bm2[n];
#pragma unroll
    for (int r = 0; r < 8; ++r) yv[t][r] = c.f[r] + bb + xv[t][r];
  }
  {
    float s[8], ss[8];
#pragma unroll
    for (int r = 0; r < 8; ++r) {
      s[r] = 0.f; ss[r] = 0.f;
#pragma unroll
      for (int t = 0; t < 8; ++t) { s[r] += yv[t][r]; ss[r] += yv[t][r] * yv[t][r]; }
    }
#pragma unroll
    for (int off = 1; off < 16; off <<= 1) {
#pragma unroll
      for (int r = 0; r < 8; ++r) {
        s[r]  += __shfl_xor(s[r], off, 32);
        ss[r] += __shfl_xor(ss[r], off, 32);
      }
    }
    float mu[8], rs[8];
#pragma unroll
    for (int r = 0; r < 8; ++r) {
      mu[r] = s[r] * (1.f / HD);
      float var = ss[r] * (1.f / HD) - mu[r] * mu[r];
      rs[r] = rsqrtf(var + 1e-5f);
    }
#pragma unroll
    for (int t = 0; t < 8; ++t) {
      int n = t * 16 + lrow;
      float gg = g2[n], bb = b2[n];
#pragma unroll
      for (int r = 0; r < 8; ++r)
        out[(size_t)(n0 + mbase + r) * HD + n] = (yv[t][r] - mu[r]) * rs[r] * gg + bb;
    }
  }
}

__global__ void copyvec_kernel(const float* __restrict__ pv, float* __restrict__ out) {
  int i = blockIdx.x * blockDim.x + threadIdx.x;
  if (i < N_PROT * 12) out[(size_t)N_PROT * HD + i] = pv[i];
}

// ------------------------------------------------------------------- launcher

extern "C" void kernel_launch(void* const* d_in, const int* in_sizes, int n_in,
                              void* d_out, int out_size, void* d_ws, size_t ws_size,
                              hipStream_t stream) {
  const float* src_feats = (const float*)d_in[0];
  const float* prot      = (const float*)d_in[1];
  const float* pvec      = (const float*)d_in[2];
  const int*   eidx_pp   = (const int*)d_in[3];
  const float* lig       = (const float*)d_in[4];
  const float* eattr     = (const float*)d_in[5];
  const int*   eidx_lp   = (const int*)d_in[6];
  const float* Watt0 = (const float*)d_in[7];
  const float* batt0 = (const float*)d_in[8];
  const float* a0    = (const float*)d_in[9];
  const float* Wval0 = (const float*)d_in[10];
  const float* bval0 = (const float*)d_in[11];
  const float* Watt1 = (const float*)d_in[12];
  const float* batt1 = (const float*)d_in[13];
  const float* a1    = (const float*)d_in[14];
  const float* Wval1 = (const float*)d_in[15];
  const float* bval1 = (const float*)d_in[16];
  const float* Wh1 = (const float*)d_in[17];
  const float* bh1 = (const float*)d_in[18];
  const float* Wh2 = (const float*)d_in[19];
  const float* bh2 = (const float*)d_in[20];
  const float* g1  = (const float*)d_in[21];
  const float* b1  = (const float*)d_in[22];
  const float* Wm1 = (const float*)d_in[23];
  const float* bm1 = (const float*)d_in[24];
  const float* Wm2 = (const float*)d_in[25];
  const float* bm2 = (const float*)d_in[26];
  const float* g2  = (const float*)d_in[27];
  const float* b2  = (const float*)d_in[28];
  float* out = (float*)d_out;

  // Workspace layout (byte offsets, each 256B-aligned).
  char* w = (char*)d_ws;
  __bf16* Watt0T = (__bf16*)(w + 0);
  __bf16* Wval0T = (__bf16*)(w + 147456);
  __bf16* Watt1T = (__bf16*)(w + 294912);
  __bf16* Wval1T = (__bf16*)(w + 393216);
  __bf16* Wh1T   = (__bf16*)(w + 491520);
  __bf16* Wh2T   = (__bf16*)(w + 524288);
  __bf16* Wm1T   = (__bf16*)(w + 557056);
  __bf16* Wm2T   = (__bf16*)(w + 688128);
  __bf16*  valb   = (__bf16*)(w + 819200);     // (E_PP+E_LP) x 128 bf16
  float*   logitb = (float*)(w + 63733760);    // (E_PP+E_LP) x 4 f32
  unsigned* maxb  = (unsigned*)(w + 67665920); // N_PROT x 4 ordered-int
  float*   sumb   = (float*)(w + 67796992);    // N_PROT x 4 f32
  float*   attnb  = (float*)(w + 67928064);    // N_PROT x 128 f32

  init_ws_kernel<<<(N_PROT * HD + 255) / 256, 256, 0, stream>>>(attnb, sumb, maxb);

  wprep_kernel<<<(576 * 128 + 255) / 256, 256, 0, stream>>>(Watt0, Watt0T, 576, 128);
  wprep_kernel<<<(576 * 128 + 255) / 256, 256, 0, stream>>>(Wval0, Wval0T, 576, 128);
  wprep_kernel<<<(384 * 128 + 255) / 256, 256, 0, stream>>>(Watt1, Watt1T, 384, 128);
  wprep_kernel<<<(384 * 128 + 255) / 256, 256, 0, stream>>>(Wval1, Wval1T, 384, 128);
  wprep_kernel<<<(128 * 128 + 255) / 256, 256, 0, stream>>>(Wh1, Wh1T, 128, 128);
  wprep_kernel<<<(128 * 128 + 255) / 256, 256, 0, stream>>>(Wh2, Wh2T, 128, 128);
  wprep_kernel<<<(128 * 512 + 255) / 256, 256, 0, stream>>>(Wm1, Wm1T, 128, 512);
  wprep_kernel<<<(512 * 128 + 255) / 256, 256, 0, stream>>>(Wm2, Wm2T, 512, 128);

  // Edge GEMMs + logits (4 waves/block, one 16-edge tile per wave).
  edge_gat_kernel<<<(E_PP_N / 16) / 4, 128, 0, stream>>>(
      src_feats, prot, nullptr, eidx_pp, Watt0T, batt0, a0, Wval0T, bval0,
      valb, logitb, maxb, E_PP_N, 576, 0);
  edge_gat_kernel<<<(E_LP_N / 16) / 4, 128, 0, stream>>>(
      lig, prot, eattr, eidx_lp, Watt1T, batt1, a1, Wval1T, bval1,
      valb + (size_t)E_PP_N * HD, logitb + (size_t)E_PP_N * NHEAD, maxb,
      E_LP_N, 384, 1);

  softmax_pass_kernel<<<((E_PP_N + E_LP_N) * NHEAD + 255) / 256, 256, 0, stream>>>(
      eidx_pp, eidx_lp, logitb, maxb, sumb);

  aggregate_kernel<<<E_PP_N + E_LP_N, 128, 0, stream>>>(
      eidx_pp, eidx_lp, logitb, sumb, valb, attnb);

  node_update_kernel<<<N_PROT / 16, 32, 0, stream>>>(
      attnb, prot, Wh1T, bh1, Wh2T, bh2, g1, b1, Wm1T, bm1, Wm2T, bm2, g2, b2, out);

  copyvec_kernel<<<(N_PROT * 12 + 255) / 256, 256, 0, stream>>>(pvec, out);

  (void)in_sizes; (void)n_in; (void)out_size; (void)ws_size;
}
